// CausalSelfAttention_68307159875607
// MI455X (gfx1250) — matrix operand order repaired and checked by reading, not compiled, not run on hardware
//
#include <hip/hip_runtime.h>
#include <math.h>
#include <stdint.h>

// ---------------------------------------------------------------------------
// CDNA5 (gfx1250) causal GQA attention layer.
// bf16 WMMA for all matmuls; TDM (tensor_load_to_lds) double-buffered GEMM
// staging; one-time f32->bf16 conversion of all matrices.
// ---------------------------------------------------------------------------

typedef __attribute__((ext_vector_type(16))) __bf16 v16bf;
typedef __attribute__((ext_vector_type(8)))  float  v8f;
typedef unsigned int u32x4 __attribute__((ext_vector_type(4)));
typedef int          i32x8 __attribute__((ext_vector_type(8)));
typedef int          i32x4 __attribute__((ext_vector_type(4)));

#define WMMA_BF16 __builtin_amdgcn_wmma_f32_16x16x32_bf16

#if defined(__has_builtin)
#if __has_builtin(__builtin_amdgcn_tensor_load_to_lds) && __has_builtin(__builtin_amdgcn_s_wait_tensorcnt)
#define HAVE_TDM 1
#endif
#endif

constexpr int Bc   = 2;
constexpr int Tc   = 2048;
constexpr int DIMc = 2048;
constexpr int Hc   = 16;
constexpr int HKVc = 4;
constexpr int Dc   = 128;
constexpr int Gc   = Hc / HKVc;   // 4
constexpr int HALFc = Dc / 2;     // 64

// LDS tile geometry: 128 rows x 32 bf16, padded to 34 elements/row (68B,
// coprime with the 64 LDS banks). One buffer = A-tile + B-tile; two buffers.
constexpr int ROWE  = 34;               // elements per padded LDS row
constexpr int TILEE = 128 * ROWE;       // elements per tile (4352)
constexpr int BUFE  = 2 * TILEE;        // elements per double-buffer slot
constexpr int TILEB = TILEE * 2;        // bytes per tile (8704)
constexpr int LDSB  = 2 * BUFE * 2;     // total dynamic LDS bytes (34816)

// A-matrix (16x32 bf16) per-lane K offset for element e of the v16bf fragment.
__device__ __forceinline__ int a_koff(int e, int hl) {
  return hl * 16 + e;
}

#ifdef HAVE_TDM
// Issue a TDM 2-D tile load: 128 rows x 32 bf16 from row-major src (row
// stride = K elements) into LDS at byte offset lds_off, +4B pad per 64B row.
__device__ __forceinline__ void tdm_load_tile(const __bf16* src, uint32_t lds_off,
                                              int K) {
  uint64_t ga = (uint64_t)(uintptr_t)src;
  u32x4 g0;
  g0[0] = 1u;                                   // count=1, user descriptor
  g0[1] = lds_off;                              // lds_addr
  g0[2] = (uint32_t)ga;                         // global_addr[31:0]
  g0[3] = ((uint32_t)(ga >> 32) & 0x1FFFFFFu) | (2u << 30); // addr[56:32]|type=2
  uint32_t d0 = (1u << 16)      // data_size = 1 (2 bytes)
              | (1u << 20)      // pad_enable
              | (3u << 22)      // pad_interval: 16 DWORDs (64B)
              | (0u << 25);     // pad_amount: 1 DWORD (4B)
  uint32_t td0 = (uint32_t)K;   // tensor_dim0 (elements)
  uint32_t td1 = 128u;          // tensor_dim1 (rows from tile start)
  uint64_t st0 = (uint64_t)K;   // tensor_dim0_stride (elements)
  i32x8 g1;
  g1[0] = (int)d0;
  g1[1] = (int)((td0 & 0xFFFFu) << 16);                  // tensor_dim0[15:0]
  g1[2] = (int)((td0 >> 16) | ((td1 & 0xFFFFu) << 16));  // td0[31:16]|td1[15:0]
  g1[3] = (int)((td1 >> 16) | (32u << 16));              // td1[31:16]|tile_dim0
  g1[4] = (int)(128u);                                   // tile_dim1 | tile_dim2<<16
  g1[5] = (int)(uint32_t)(st0 & 0xFFFFFFFFu);            // stride0[31:0]
  g1[6] = (int)(uint32_t)(st0 >> 32);                    // stride0[47:32]|stride1lo
  g1[7] = 0;
  i32x4 z4 = {0, 0, 0, 0};
  i32x8 z8 = {0, 0, 0, 0, 0, 0, 0, 0};
  // 6-arg toolchain variant: (g0, g1, g2, g3, g2b/g3b ext, cpol)
  __builtin_amdgcn_tensor_load_to_lds(g0, g1, z4, z4, z8, 0);
}
#endif

// ---------------------------------------------------------------------------
// One-time f32 -> bf16 conversion (vectorized).
// ---------------------------------------------------------------------------
__global__ void cvt_f32_bf16(const float* __restrict__ src,
                             __bf16* __restrict__ dst, int n) {
  int i = (blockIdx.x * blockDim.x + threadIdx.x) * 4;
  if (i + 3 < n) {
    float4 f = *reinterpret_cast<const float4*>(src + i);
    dst[i + 0] = (__bf16)f.x;
    dst[i + 1] = (__bf16)f.y;
    dst[i + 2] = (__bf16)f.z;
    dst[i + 3] = (__bf16)f.w;
  }
}

// ---------------------------------------------------------------------------
// GEMM: C[M,N] (f32) = A[M,K] (bf16, row-major) @ W[N,K]^T (bf16, row-major)
// 256 threads (8 waves), 128x128 block tile, BK=32. Wave grid 4x2; each wave
// computes 32x64 = 8 WMMAs per K-step from 2 A-frags + 4 B-frags.
// Tiles staged into LDS by the TDM (double-buffered, DMA overlaps WMMA).
// ---------------------------------------------------------------------------
__global__ __launch_bounds__(256)
void gemm_bf16_wmma(const __bf16* __restrict__ A, const __bf16* __restrict__ W,
                    float* __restrict__ C, int M, int N, int K) {
  extern __shared__ __bf16 smem[];

  const int tid  = threadIdx.x;
  const int lane = tid & 31;
  const int hl   = lane >> 4;
  const int lh   = lane & 15;
  const int w    = tid >> 5;
  const int wm   = (w >> 1) * 32;
  const int wn   = (w & 1) * 64;
  const int mBase = blockIdx.y * 128;
  const int nBase = blockIdx.x * 128;

  v8f acc[2][4];
#pragma unroll
  for (int mi = 0; mi < 2; ++mi)
#pragma unroll
    for (int ni = 0; ni < 4; ++ni)
#pragma unroll
      for (int r = 0; r < 8; ++r) acc[mi][ni][r] = 0.f;

#ifdef HAVE_TDM
  // ---- prologue: DMA tile 0 into buffer 0 (wave 0 issues)
  if (w == 0) {
    tdm_load_tile(&A[(size_t)mBase * K], 0u, K);
    tdm_load_tile(&W[(size_t)nBase * K], (uint32_t)TILEB, K);
  }
  int buf = 0;
  for (int k0 = 0; k0 < K; k0 += 32) {
    if (w == 0) __builtin_amdgcn_s_wait_tensorcnt(0);
    __syncthreads();                       // tile k0 visible to all waves
    if (w == 0 && k0 + 32 < K) {           // DMA tile k0+32 into other buffer
      int nb = buf ^ 1;
      tdm_load_tile(&A[(size_t)mBase * K + k0 + 32], (uint32_t)(nb * 2 * TILEB), K);
      tdm_load_tile(&W[(size_t)nBase * K + k0 + 32], (uint32_t)(nb * 2 * TILEB + TILEB), K);
    }
    const __bf16* As = smem + buf * BUFE;
    const __bf16* Bs = As + TILEE;
#else
  int buf = 0;
  const int srow = tid >> 1;
  const int scol = (tid & 1) * 16;
  for (int k0 = 0; k0 < K; k0 += 32) {
    if (k0 + 32 < K) {
      __builtin_prefetch(&A[(size_t)(mBase + srow) * K + k0 + 32 + scol], 0, 3);
      __builtin_prefetch(&W[(size_t)(nBase + srow) * K + k0 + 32 + scol], 0, 3);
    }
#pragma unroll
    for (int j = 0; j < 16; ++j) {
      smem[srow * ROWE + scol + j]         = A[(size_t)(mBase + srow) * K + k0 + scol + j];
      smem[TILEE + srow * ROWE + scol + j] = W[(size_t)(nBase + srow) * K + k0 + scol + j];
    }
    __syncthreads();
    const __bf16* As = smem;
    const __bf16* Bs = smem + TILEE;
#endif

    v16bf a[2], b[4];
#pragma unroll
    for (int mi = 0; mi < 2; ++mi)
#pragma unroll
      for (int e = 0; e < 16; ++e)
        a[mi][e] = As[(wm + mi * 16 + lh) * ROWE + a_koff(e, hl)];
#pragma unroll
    for (int ni = 0; ni < 4; ++ni)
#pragma unroll
      for (int e = 0; e < 16; ++e)
        b[ni][e] = Bs[(wn + ni * 16 + lh) * ROWE + hl * 16 + e];

#pragma unroll
    for (int mi = 0; mi < 2; ++mi)
#pragma unroll
      for (int ni = 0; ni < 4; ++ni)
        acc[mi][ni] = WMMA_BF16(false, a[mi], false, b[ni], (short)0,
                                acc[mi][ni], false, false);
#ifdef HAVE_TDM
    buf ^= 1;
#else
    __syncthreads();
#endif
  }

#pragma unroll
  for (int mi = 0; mi < 2; ++mi)
#pragma unroll
    for (int ni = 0; ni < 4; ++ni)
#pragma unroll
      for (int r = 0; r < 8; ++r)
        C[(size_t)(mBase + wm + mi * 16 + r + 8 * hl) * N
          + nBase + wn + ni * 16 + lh] = acc[mi][ni][r];
}

// ---------------------------------------------------------------------------
// Fused RMSNorm + NTK-RoPE; reads f32 GEMM output, writes bf16 for attention.
// gain * 1/sqrt(D) folded into q via outScale.
// ---------------------------------------------------------------------------
__global__ void normrope(const float* __restrict__ src, __bf16* __restrict__ dst,
                         int nheads, const float* __restrict__ gain,
                         float outScale) {
  int idx = blockIdx.x * blockDim.x + threadIdx.x;
  int total = Bc * Tc * nheads;
  if (idx >= total) return;
  int h   = idx % nheads;
  int row = idx / nheads;
  int t   = row % Tc;
  size_t off = (size_t)row * nheads * Dc + (size_t)h * Dc;
  const float* p = src + off;
  __bf16* o = dst + off;

  float ss = 0.f;
  for (int i = 0; i < Dc; ++i) { float f = p[i]; ss += f * f; }
  float rms = rsqrtf(ss * (1.0f / Dc) + 1.1920929e-7f);

  float g = outScale * (gain ? gain[h] : 1.0f);
  // T > TRAIN_LEN -> NTK scaling: base = 10000 * (T/1024)^(D/(D-2))
  float base = 10000.0f * __powf((float)Tc / 1024.0f, (float)Dc / (float)(Dc - 2));
  for (int i = 0; i < HALFc; ++i) {
    float invf = __powf(base, -2.0f * (float)i / (float)Dc);
    float ang = (float)t * invf;
    float c, s;
    __sincosf(ang, &c, &s);
    float x1 = p[i] * rms, x2 = p[i + HALFc] * rms;
    o[i]         = (__bf16)((x1 * c + x2 * s) * g);
    o[i + HALFc] = (__bf16)((-x1 * s + x2 * c) * g);
  }
}

// ---------------------------------------------------------------------------
// Flash attention: one wave per (b, h, 16-row q tile). bf16 Q/K/V in, bf16 Y
// out. Online softmax; 8 score WMMAs + 8 PV WMMAs per 32-key tile.
// ---------------------------------------------------------------------------
__global__ __launch_bounds__(32)
void attn_wmma(const __bf16* __restrict__ Q, const __bf16* __restrict__ K,
               const __bf16* __restrict__ V, __bf16* __restrict__ Y) {
  __shared__ __bf16 Pl[16 * 32];

  const int lane = threadIdx.x & 31;
  const int hl = lane >> 4, lh = lane & 15;
  const int NT = Tc / 16;
  int bid = blockIdx.x;
  const int qt = bid % NT; bid /= NT;
  const int h  = bid % Hc;
  const int b  = bid / Hc;
  const int hk = h / Gc;
  const int qt16 = qt * 16;

  v16bf aQ[4];
#pragma unroll
  for (int c = 0; c < 4; ++c)
#pragma unroll
    for (int e = 0; e < 16; ++e)
      aQ[c][e] = Q[((size_t)(b * Tc + qt16 + lh) * Hc + h) * Dc
                   + c * 32 + a_koff(e, hl)];

  v8f acc[8];
#pragma unroll
  for (int i = 0; i < 8; ++i)
#pragma unroll
    for (int r = 0; r < 8; ++r) acc[i][r] = 0.f;

  float mrow[8], lsum[8];
#pragma unroll
  for (int r = 0; r < 8; ++r) { mrow[r] = -1e30f; lsum[r] = 0.f; }

  for (int s0 = 0; s0 <= qt16 + 15; s0 += 32) {
    if (s0 + 32 <= qt16 + 15) {
      __builtin_prefetch(&K[((size_t)(b * Tc + s0 + 32 + lane) * HKVc + hk) * Dc], 0, 3);
      __builtin_prefetch(&V[((size_t)(b * Tc + s0 + 32 + lane) * HKVc + hk) * Dc], 0, 3);
    }
    // ---- scores: two 16x16 tiles over 32 keys, K-dim = D via 4 WMMAs each
    v8f S0, S1;
#pragma unroll
    for (int r = 0; r < 8; ++r) { S0[r] = 0.f; S1[r] = 0.f; }
#pragma unroll
    for (int c = 0; c < 4; ++c) {
      v16bf b0, b1;
#pragma unroll
      for (int e = 0; e < 16; ++e) {
        int kd = c * 32 + hl * 16 + e;
        b0[e] = K[((size_t)(b * Tc + s0 + lh) * HKVc + hk) * Dc + kd];
        b1[e] = K[((size_t)(b * Tc + s0 + 16 + lh) * HKVc + hk) * Dc + kd];
      }
      S0 = WMMA_BF16(false, aQ[c], false, b0, (short)0, S0, false, false);
      S1 = WMMA_BF16(false, aQ[c], false, b1, (short)0, S1, false, false);
    }

    // ---- online softmax (rows split across 16-lane halves per C layout)
    float p0[8], p1[8], alpha[8];
#pragma unroll
    for (int r = 0; r < 8; ++r) {
      int rowg = qt16 + r + 8 * hl;
      float s0v = (s0 + lh      <= rowg) ? S0[r] : -1e30f;
      float s1v = (s0 + 16 + lh <= rowg) ? S1[r] : -1e30f;
      float mx = fmaxf(s0v, s1v);
#pragma unroll
      for (int msk = 1; msk <= 8; msk <<= 1)
        mx = fmaxf(mx, __shfl_xor(mx, msk, 32));
      float mnew = fmaxf(mrow[r], mx);
      alpha[r] = __expf(mrow[r] - mnew);
      mrow[r] = mnew;
      p0[r] = __expf(s0v - mnew);
      p1[r] = __expf(s1v - mnew);
      float rs = p0[r] + p1[r];
#pragma unroll
      for (int msk = 1; msk <= 8; msk <<= 1)
        rs += __shfl_xor(rs, msk, 32);
      lsum[r] = lsum[r] * alpha[r] + rs;
    }
#pragma unroll
    for (int dt = 0; dt < 8; ++dt)
#pragma unroll
      for (int r = 0; r < 8; ++r) acc[dt][r] *= alpha[r];

    // ---- re-swizzle P (C layout -> A layout) through LDS
#pragma unroll
    for (int r = 0; r < 8; ++r) {
      Pl[(r + 8 * hl) * 32 + lh]      = (__bf16)p0[r];
      Pl[(r + 8 * hl) * 32 + 16 + lh] = (__bf16)p1[r];
    }
    __syncthreads();
    v16bf aP;
#pragma unroll
    for (int e = 0; e < 16; ++e) aP[e] = Pl[lh * 32 + a_koff(e, hl)];
    __syncthreads();

    // ---- PV: 8 d-tiles of 16 columns
#pragma unroll
    for (int dt = 0; dt < 8; ++dt) {
      v16bf bV;
#pragma unroll
      for (int e = 0; e < 16; ++e)
        bV[e] = V[((size_t)(b * Tc + s0 + hl * 16 + e) * HKVc + hk) * Dc
                  + dt * 16 + lh];
      acc[dt] = WMMA_BF16(false, aP, false, bV, (short)0, acc[dt], false, false);
    }
  }

#pragma unroll
  for (int dt = 0; dt < 8; ++dt)
#pragma unroll
    for (int r = 0; r < 8; ++r)
      Y[((size_t)(b * Tc + qt16 + r + 8 * hl) * Hc + h) * Dc + dt * 16 + lh] =
          (__bf16)(acc[dt][r] / lsum[r]);
}

// ---------------------------------------------------------------------------
// Host-side launch
// ---------------------------------------------------------------------------
extern "C" void kernel_launch(void* const* d_in, const int* in_sizes, int n_in,
                              void* d_out, int out_size, void* d_ws, size_t ws_size,
                              hipStream_t stream) {
  const float* x     = (const float*)d_in[0];
  const float* Wq    = (const float*)d_in[1];
  const float* Wk    = (const float*)d_in[2];
  const float* Wv    = (const float*)d_in[3];
  const float* Wproj = (const float*)d_in[4];
  const float* qgain = (const float*)d_in[5];
  float* out = (float*)d_out;

  char* ws = (char*)d_ws;
  float*  qf  = (float*)(ws + 0);           // 32MB f32 q
  float*  kf  = (float*)(ws + 33554432);    //  8MB f32 k
  float*  vf  = (float*)(ws + 41943040);    //  8MB f32 v
  __bf16* xb  = (__bf16*)(ws + 50331648);   // 16MB bf16 x
  __bf16* wqb = (__bf16*)(ws + 67108864);   //  8MB bf16 Wq
  __bf16* wkb = (__bf16*)(ws + 75497472);   //  2MB bf16 Wk
  __bf16* wvb = (__bf16*)(ws + 77594624);   //  2MB bf16 Wv
  __bf16* wpb = (__bf16*)(ws + 79691776);   //  8MB bf16 Wproj
  __bf16* qb  = (__bf16*)(ws + 88080384);   // 16MB bf16 q (norm+rope)
  __bf16* kb  = (__bf16*)(ws + 104857600);  //  4MB bf16 k (norm+rope)
  __bf16* vb  = (__bf16*)(ws + 109051904);  //  4MB bf16 v
  __bf16* yb  = (__bf16*)(ws + 113246208);  // 16MB bf16 attention out

  const int M = Bc * Tc;   // 4096
  dim3 blk(256);
  auto cvt = [&](const float* s, __bf16* d, int n) {
    cvt_f32_bf16<<<(n / 4 + 255) / 256, 256, 0, stream>>>(s, d, n);
  };

  // one-time bf16 conversions
  cvt(x, xb, M * DIMc);
  cvt(Wq, wqb, DIMc * DIMc);
  cvt(Wk, wkb, HKVc * Dc * DIMc);
  cvt(Wv, wvb, HKVc * Dc * DIMc);
  cvt(Wproj, wpb, DIMc * DIMc);

  // QKV projections (bf16 in, f32 out)
  gemm_bf16_wmma<<<dim3(DIMc / 128, M / 128), blk, LDSB, stream>>>(xb, wqb, qf, M, DIMc, DIMc);
  gemm_bf16_wmma<<<dim3((HKVc * Dc) / 128, M / 128), blk, LDSB, stream>>>(xb, wkb, kf, M, HKVc * Dc, DIMc);
  gemm_bf16_wmma<<<dim3((HKVc * Dc) / 128, M / 128), blk, LDSB, stream>>>(xb, wvb, vf, M, HKVc * Dc, DIMc);

  // norm + rope (+gain/sqrt(D) for q), f32 -> bf16
  int tq = Bc * Tc * Hc;
  normrope<<<(tq + 255) / 256, 256, 0, stream>>>(qf, qb, Hc, qgain, 0.08838834764831845f);
  int tk = Bc * Tc * HKVc;
  normrope<<<(tk + 255) / 256, 256, 0, stream>>>(kf, kb, HKVc, nullptr, 1.0f);
  cvt(vf, vb, Bc * Tc * HKVc * Dc);

  attn_wmma<<<Bc * Hc * (Tc / 16), 32, 0, stream>>>(qb, kb, vb, yb);

  // output projection
  gemm_bf16_wmma<<<dim3(DIMc / 128, M / 128), blk, LDSB, stream>>>(yb, wpb, out, M, DIMc, DIMc);
}
